// GroupedQueryAttention_1657857376336
// MI455X (gfx1250) — matrix operand order repaired and checked by reading, not compiled, not run on hardware
//
#include <hip/hip_runtime.h>
#include <cstddef>

// ---------------------------------------------------------------------------
// CDNA5 (gfx1250) fused GQA: qkv GEMM -> RoPE -> windowed attention -> out GEMM
// Matmuls: v_wmma_f32_16x16x32_bf16 (fp32 downcast to bf16, f32 accumulate).
// Attention K/V tiles: TDM tensor_load_to_lds (DMA + LDS padding) when the
// toolchain exposes the builtin; cooperative vector loads otherwise.
// ---------------------------------------------------------------------------

#define SEQ   2048
#define HID   2880
#define NH    64
#define NKV   8
#define HD    64
#define WINSZ 128
#define QKVN  ((NH + 2 * NKV) * HD)   // 5120
#define AOUTK (NH * HD)               // 4096

typedef __attribute__((ext_vector_type(16))) __bf16 v16bf;
typedef __attribute__((ext_vector_type(8)))  float  v8f;
typedef unsigned int u32x4 __attribute__((ext_vector_type(4)));
typedef int          i32x4 __attribute__((ext_vector_type(4)));
typedef int          i32x8 __attribute__((ext_vector_type(8)));

#if defined(__AMDGCN__) && __has_builtin(__builtin_amdgcn_tensor_load_to_lds)
#define HAVE_TDM 1
#else
#define HAVE_TDM 0
#endif

struct __align__(16) U4 { unsigned int a, b, c, d; };
struct __align__(16) F4 { float x, y, z, w; };

union Frag { v16bf v; U4 q[2]; };

__device__ __forceinline__ unsigned short f2bf(float f) {
    unsigned int u = __float_as_uint(f);
    u += 0x7FFFu + ((u >> 16) & 1u);          // round-to-nearest-even
    return (unsigned short)(u >> 16);
}
__device__ __forceinline__ float bf2f(unsigned short h) {
    return __uint_as_float(((unsigned int)h) << 16);
}

__device__ __forceinline__ v8f wmma_bf16(const Frag& A, const Frag& B, v8f C) {
    return __builtin_amdgcn_wmma_f32_16x16x32_bf16(false, A.v, false, B.v,
                                                   (short)0, C, false, false);
}

#if HAVE_TDM
// 2-D TDM load: tile (tile1 rows x tile0 elems) of bf16 from a row-major
// tensor (row length dim0, stride stride0 elems) into LDS at lds_addr.
// g1w0 carries workgroup_mask=0 | data_size=2B | pad_enable/interval/amount.
__device__ __forceinline__ void tdm_load_2d(
    unsigned lds_addr, const void* gptr, unsigned g1w0,
    unsigned dim0, unsigned dim1, unsigned tile0, unsigned tile1,
    unsigned stride0)
{
    const unsigned long long ga = (unsigned long long)(uintptr_t)gptr;
    u32x4 g0;
    g0[0] = 1u;                                             // count=1, user
    g0[1] = lds_addr;                                       // lds_addr[31:0]
    g0[2] = (unsigned)ga;                                   // global_addr lo
    g0[3] = ((unsigned)(ga >> 32) & 0x01FFFFFFu) | 0x80000000u; // hi | type=2
    i32x8 g1;
    g1[0] = (int)g1w0;
    g1[1] = (int)(dim0 << 16);                    // atomic_barrier=0 | dim0.lo
    g1[2] = (int)((dim0 >> 16) | (dim1 << 16));   // dim0.hi | dim1.lo
    g1[3] = (int)((dim1 >> 16) | (tile0 << 16));  // dim1.hi | tile_dim0
    g1[4] = (int)(tile1 & 0xFFFFu);               // tile_dim1 | tile_dim2=0
    g1[5] = (int)stride0;                         // dim0_stride lo
    g1[6] = 0;                                    // stride hi | dim1_stride lo
    g1[7] = 0;
    const i32x4 z4 = {0, 0, 0, 0};
#if __clang_major__ >= 23
    const i32x8 z8 = {0, 0, 0, 0, 0, 0, 0, 0};
    __builtin_amdgcn_tensor_load_to_lds(g0, g1, z4, z4, z8, 0);
#else
    __builtin_amdgcn_tensor_load_to_lds(g0, g1, z4, z4, 0);
#endif
}
#endif

// ---- LDS tile loaders (convert to bf16 on the way in) ----------------------
template <typename TA>
__device__ __forceinline__ void loadcvt16(const TA* g, unsigned short* s);

template <>
__device__ __forceinline__ void loadcvt16<float>(const float* g, unsigned short* s) {
    const F4* g4 = (const F4*)g;
#pragma unroll
    for (int i = 0; i < 4; ++i) {
        F4 v = g4[i];
        s[i * 4 + 0] = f2bf(v.x); s[i * 4 + 1] = f2bf(v.y);
        s[i * 4 + 2] = f2bf(v.z); s[i * 4 + 3] = f2bf(v.w);
    }
}
template <>
__device__ __forceinline__ void loadcvt16<unsigned short>(const unsigned short* g,
                                                          unsigned short* s) {
    const U4* g4 = (const U4*)g;
    *(U4*)s       = g4[0];
    *(U4*)(s + 8) = g4[1];
}

__device__ __forceinline__ void loadcvt8_f32(const float* g, unsigned short* s) {
    const F4* g4 = (const F4*)g;
    F4 a = g4[0], b = g4[1];
    s[0] = f2bf(a.x); s[1] = f2bf(a.y); s[2] = f2bf(a.z); s[3] = f2bf(a.w);
    s[4] = f2bf(b.x); s[5] = f2bf(b.y); s[6] = f2bf(b.z); s[7] = f2bf(b.w);
}

template <typename TC>
__device__ __forceinline__ void storeC(TC* p, float v);
template <> __device__ __forceinline__ void storeC<float>(float* p, float v) { *p = v; }
template <> __device__ __forceinline__ void storeC<unsigned short>(unsigned short* p, float v) { *p = f2bf(v); }

// ---------------------------------------------------------------------------
// C[M][N] = A[M][K] @ B[N][K]^T + bias.  Block tile 128 x BN, 8 waves (4x2),
// wave tile 32 x (BN/2), K-step 32, bf16 WMMA with f32 accumulation.
// ---------------------------------------------------------------------------
template <int BN, typename TA, typename TC>
__global__ __launch_bounds__(256) void gemm_bias_kernel(
    const TA* __restrict__ A, int lda,
    const float* __restrict__ Bw, int ldb,     // row-major [N][K]
    const float* __restrict__ bias,
    TC* __restrict__ C, int ldc,
    int M, int N, int K)
{
    constexpr int BM = 128, BK = 32;
    constexpr int WN = BN / 2;                 // per-wave N extent
    constexpr int NT = WN / 16;                // N subtiles per wave
    constexpr int LDS_S = BK + 8;              // padded stride (bank spread)
    __shared__ __align__(16) unsigned short sA[BM * LDS_S];
    __shared__ __align__(16) unsigned short sB[BN * LDS_S];

    const int tid  = threadIdx.x;
    const int lane = tid & 31;
    const int wave = tid >> 5;
    const int wm = wave >> 1, wn = wave & 1;   // 4 x 2 wave grid
    const int m0 = blockIdx.y * BM;
    const int n0 = blockIdx.x * BN;

    v8f acc[2][NT];
    const v8f zf = {0.f, 0.f, 0.f, 0.f, 0.f, 0.f, 0.f, 0.f};
#pragma unroll
    for (int i = 0; i < 2; ++i)
#pragma unroll
        for (int j = 0; j < NT; ++j) acc[i][j] = zf;

    const int ra = tid >> 1, ca = (tid & 1) * 16;   // A tile: 128 x 32
    // B tile: BN x 32 -> 16 (BN=128) or 8 (BN=64) elements per thread
    const int rb = (BN == 128) ? (tid >> 1) : (tid >> 2);
    const int cb = (BN == 128) ? ((tid & 1) * 16) : ((tid & 3) * 8);

    for (int k0 = 0; k0 < K; k0 += BK) {
        loadcvt16<TA>(A + (size_t)(m0 + ra) * lda + k0 + ca, &sA[ra * LDS_S + ca]);
        if (BN == 128)
            loadcvt16<float>(Bw + (size_t)(n0 + rb) * ldb + k0 + cb, &sB[rb * LDS_S + cb]);
        else
            loadcvt8_f32(Bw + (size_t)(n0 + rb) * ldb + k0 + cb, &sB[rb * LDS_S + cb]);
        if (k0 + BK < K) {                      // global_prefetch next K tiles
            __builtin_prefetch(A  + (size_t)(m0 + ra) * lda + k0 + BK + ca, 0, 1);
            __builtin_prefetch(Bw + (size_t)(n0 + rb) * ldb + k0 + BK + cb, 0, 1);
        }
        __syncthreads();

        Frag aF[2], bF[NT];
        const int amb = wm * 32 + (lane & 15);
        const int kb  = (lane < 16) ? 0 : 8;
#pragma unroll
        for (int mt = 0; mt < 2; ++mt) {
            const unsigned short* p = &sA[(amb + mt * 16) * LDS_S + kb];
            aF[mt].q[0] = *(const U4*)p;
            aF[mt].q[1] = *(const U4*)(p + 16);
        }
        const int bnb = wn * WN + (lane & 15);
        const int ko  = (lane < 16) ? 0 : 8;
#pragma unroll
        for (int nt = 0; nt < NT; ++nt) {
            const unsigned short* p = &sB[(bnb + nt * 16) * LDS_S + ko];
            bF[nt].q[0] = *(const U4*)p;
            bF[nt].q[1] = *(const U4*)(p + 16);
        }
#pragma unroll
        for (int mt = 0; mt < 2; ++mt)
#pragma unroll
            for (int nt = 0; nt < NT; ++nt)
                acc[mt][nt] = wmma_bf16(aF[mt], bF[nt], acc[mt][nt]);
        __syncthreads();
    }

#pragma unroll
    for (int mt = 0; mt < 2; ++mt)
#pragma unroll
        for (int nt = 0; nt < NT; ++nt) {
            const int n  = n0 + wn * WN + nt * 16 + (lane & 15);
            const float bv = bias[n];
#pragma unroll
            for (int r = 0; r < 8; ++r) {
                const int m = m0 + wm * 32 + mt * 16 + r + ((lane >> 4) << 3);
                storeC<TC>(&C[(size_t)m * ldc + n], acc[mt][nt][r] + bv);
            }
        }
}

// ---------------------------------------------------------------------------
// RoPE + head reshape: qkv bf16 [S][5120] -> q[64][S][64], k[8][S][64],
// vT[8][64][S] (V transposed so P@V B-fragments are contiguous).
// ---------------------------------------------------------------------------
__global__ __launch_bounds__(256) void rope_reshape_kernel(
    const unsigned short* __restrict__ qkv,
    unsigned short* __restrict__ qh,
    unsigned short* __restrict__ kh,
    unsigned short* __restrict__ vT)
{
    const int idx  = blockIdx.x * 256 + threadIdx.x;   // S * 80 * 32 total
    const int s    = idx / 2560;
    const int rem  = idx - s * 2560;
    const int head = rem >> 5;
    const int d2   = rem & 31;
    if (s >= SEQ) return;

    const unsigned short* row = qkv + (size_t)s * QKVN + head * HD;
    float x1 = bf2f(row[d2]);
    float x2 = bf2f(row[d2 + 32]);
    if (head < NH + NKV) {   // rope applies to q and k
        const float inv_freq = __expf(-((float)(2 * d2) * (1.0f / 64.0f)) * logf(150000.0f));
        const float ang = ((float)s * (1.0f / 32.0f)) * inv_freq;
        float c, sn;
        __sincosf(ang, &sn, &c);
        const float y1 = x1 * c - x2 * sn;
        const float y2 = x2 * c + x1 * sn;
        x1 = y1; x2 = y2;
    }
    const unsigned short o1 = f2bf(x1), o2 = f2bf(x2);
    if (head < NH) {
        unsigned short* q = qh + ((size_t)head * SEQ + s) * HD;
        q[d2] = o1; q[d2 + 32] = o2;
    } else if (head < NH + NKV) {
        unsigned short* k = kh + ((size_t)(head - NH) * SEQ + s) * HD;
        k[d2] = o1; k[d2 + 32] = o2;
    } else {
        unsigned short* v = vT + (size_t)(head - NH - NKV) * HD * SEQ;
        v[(size_t)d2 * SEQ + s]        = o1;
        v[(size_t)(d2 + 32) * SEQ + s] = o2;
    }
}

// ---------------------------------------------------------------------------
// Flash-style windowed attention with sinks.
// Block = 128 threads (4 waves); wave owns 16 queries; 32-key K/V tiles in
// LDS, staged by the Tensor Data Mover when available.
// ---------------------------------------------------------------------------
__global__ __launch_bounds__(128) void attn_kernel(
    const unsigned short* __restrict__ qh,   // [64][S][64]
    const unsigned short* __restrict__ kh,   // [8][S][64]
    const unsigned short* __restrict__ vT,   // [8][64][S]
    const float* __restrict__ sinks,
    unsigned short* __restrict__ attnout)    // [S][4096]
{
    constexpr int KSTR = 72, VSTR = 40, PSTR = 40;   // padded LDS strides
    __shared__ __align__(16) unsigned short sK[32 * KSTR];
    __shared__ __align__(16) unsigned short sV[64 * VSTR];
    __shared__ __align__(16) unsigned short sP[4][16 * PSTR];

    const int tid = threadIdx.x, lane = tid & 31, wave = tid >> 5;
    const int h = blockIdx.y, kvh = h >> 3;
    const int i0 = blockIdx.x * 64;
    const int iw = i0 + wave * 16;
    const float scale = 0.125f;                // 1/sqrt(64)
    const float sinkv = sinks[h];

    // Q fragments (A-layout, 16x32 per K-half of D=64)
    Frag qA[2];
    {
        const int m  = iw + (lane & 15);
        const int kb = (lane < 16) ? 0 : 8;
        const unsigned short* qr = qh + ((size_t)h * SEQ + m) * HD;
        qA[0].q[0] = *(const U4*)(qr + kb);
        qA[0].q[1] = *(const U4*)(qr + kb + 16);
        qA[1].q[0] = *(const U4*)(qr + 32 + kb);
        qA[1].q[1] = *(const U4*)(qr + 32 + kb + 16);
    }

    const v8f zf = {0.f, 0.f, 0.f, 0.f, 0.f, 0.f, 0.f, 0.f};
    v8f accO[4];
    float rowmax[8], rowsum[8];
#pragma unroll
    for (int dt = 0; dt < 4; ++dt) accO[dt] = zf;
#pragma unroll
    for (int r = 0; r < 8; ++r) { rowmax[r] = -3.0e38f; rowsum[r] = 0.f; }

    int jstart = i0 - (WINSZ - 1);
    jstart = (jstart < 0) ? 0 : (jstart & ~31);
    const int iRow0 = iw + ((lane >> 4) << 3);

    for (int jb = jstart; jb < SEQ; jb += 32) {
#if HAVE_TDM
        if (wave == 0) {
            // K tile: 32 rows x 64 bf16, row = 128B = 32 DW -> pad code 4,
            // pad 16B = 4 DW -> code 3  => LDS stride 72 elems.
            tdm_load_2d((unsigned)(uintptr_t)&sK[0],
                        kh + ((size_t)kvh * SEQ + jb) * HD,
                        0x07110000u, HD, SEQ, HD, 32, HD);
            // V tile: 64 rows x 32 bf16, row = 64B = 16 DW -> pad code 3,
            // pad 16B = 4 DW -> code 3  => LDS stride 40 elems.
            tdm_load_2d((unsigned)(uintptr_t)&sV[0],
                        vT + (size_t)kvh * HD * SEQ + jb,
                        0x06D10000u, SEQ, HD, 32, HD, SEQ);
            __builtin_amdgcn_s_wait_tensorcnt(0);
        }
#else
        {   // cooperative K/V tile load (bf16, already converted)
            const int jr = tid >> 2, cc = (tid & 3) * 16;
            const U4* ks = (const U4*)(kh + ((size_t)kvh * SEQ + jb + jr) * HD + cc);
            *(U4*)&sK[jr * KSTR + cc]     = ks[0];
            *(U4*)&sK[jr * KSTR + cc + 8] = ks[1];
            const int dr = tid >> 1, jc = (tid & 1) * 16;
            const U4* vs = (const U4*)(vT + ((size_t)kvh * HD + dr) * SEQ + jb + jc);
            *(U4*)&sV[dr * VSTR + jc]     = vs[0];
            *(U4*)&sV[dr * VSTR + jc + 8] = vs[1];
        }
#endif
        __syncthreads();

        // scores: q (16x64) @ kT (64x32) -> two 16x16 accumulators
        v8f accS[2];
        {
            const int ko = (lane < 16) ? 0 : 8;
#pragma unroll
            for (int g = 0; g < 2; ++g) {
                Frag b0, b1;
                const unsigned short* kp = &sK[(g * 16 + (lane & 15)) * KSTR];
                b0.q[0] = *(const U4*)(kp + ko);
                b0.q[1] = *(const U4*)(kp + ko + 16);
                b1.q[0] = *(const U4*)(kp + 32 + ko);
                b1.q[1] = *(const U4*)(kp + 32 + ko + 16);
                accS[g] = wmma_bf16(qA[0], b0, zf);
                accS[g] = wmma_bf16(qA[1], b1, accS[g]);
            }
        }

        // elementwise: scale, window mask (j >= i-127), sink on column 0
        float val[2][8];
#pragma unroll
        for (int g = 0; g < 2; ++g) {
            const int j = jb + g * 16 + (lane & 15);
#pragma unroll
            for (int r = 0; r < 8; ++r) {
                float v = accS[g][r] * scale;
                if (j < iRow0 + r - (WINSZ - 1)) v -= 10000.f;
                if (j == 0) v += sinkv;
                val[g][r] = v;
            }
        }

        // streaming softmax (rows live per half-wave: reduce across 16 lanes)
#pragma unroll
        for (int r = 0; r < 8; ++r) {
            float t = fmaxf(val[0][r], val[1][r]);
#pragma unroll
            for (int off = 1; off < 16; off <<= 1)
                t = fmaxf(t, __shfl_xor(t, off, 16));
            const float nmax  = fmaxf(rowmax[r], t);
            const float alpha = __expf(rowmax[r] - nmax);
            rowmax[r] = nmax;
            const float p0 = __expf(val[0][r] - nmax);
            const float p1 = __expf(val[1][r] - nmax);
            float s = p0 + p1;
#pragma unroll
            for (int off = 1; off < 16; off <<= 1)
                s += __shfl_xor(s, off, 16);
            rowsum[r] = rowsum[r] * alpha + s;
#pragma unroll
            for (int dt = 0; dt < 4; ++dt) accO[dt][r] *= alpha;
            // stage P (C-layout -> LDS) for A-layout reload
            const int m = r + ((lane >> 4) << 3);
            unsigned short* pw = &sP[wave][m * PSTR];
            pw[(lane & 15)]      = f2bf(p0);
            pw[16 + (lane & 15)] = f2bf(p1);
        }

        // P (16x32) @ V (32x64): 4 WMMAs
        {
            Frag pF;
            const int m  = lane & 15;
            const int kb = (lane < 16) ? 0 : 8;
            const unsigned short* pp = &sP[wave][m * PSTR + kb];
            pF.q[0] = *(const U4*)pp;
            pF.q[1] = *(const U4*)(pp + 16);
            const int jo = (lane < 16) ? 0 : 8;
#pragma unroll
            for (int dt = 0; dt < 4; ++dt) {
                Frag vF;
                const unsigned short* vp = &sV[(dt * 16 + (lane & 15)) * VSTR + jo];
                vF.q[0] = *(const U4*)vp;
                vF.q[1] = *(const U4*)(vp + 16);
                accO[dt] = wmma_bf16(pF, vF, accO[dt]);
            }
        }
        __syncthreads();
    }

    // normalize and scatter to [S][H*D] bf16
#pragma unroll
    for (int dt = 0; dt < 4; ++dt) {
        const int d = dt * 16 + (lane & 15);
#pragma unroll
        for (int r = 0; r < 8; ++r) {
            const int s = iw + r + ((lane >> 4) << 3);
            const float o = accO[dt][r] / rowsum[r];
            attnout[(size_t)s * AOUTK + h * HD + d] = f2bf(o);
        }
    }
}

// ---------------------------------------------------------------------------
extern "C" void kernel_launch(void* const* d_in, const int* in_sizes, int n_in,
                              void* d_out, int out_size, void* d_ws, size_t ws_size,
                              hipStream_t stream)
{
    (void)in_sizes; (void)n_in; (void)out_size; (void)ws_size;
    const float* hidden = (const float*)d_in[0];
    const float* qkv_w  = (const float*)d_in[1];
    const float* qkv_b  = (const float*)d_in[2];
    const float* out_w  = (const float*)d_in[3];
    const float* out_b  = (const float*)d_in[4];
    const float* sinks  = (const float*)d_in[5];
    float* out = (float*)d_out;

    unsigned short* ws      = (unsigned short*)d_ws;
    unsigned short* qkv     = ws;                                  // [S][5120]
    unsigned short* qh      = qkv + (size_t)SEQ * QKVN;            // [64][S][64]
    unsigned short* kh      = qh  + (size_t)NH * SEQ * HD;         // [8][S][64]
    unsigned short* vT      = kh  + (size_t)NKV * SEQ * HD;        // [8][64][S]
    unsigned short* attnout = vT  + (size_t)NKV * HD * SEQ;        // [S][4096]

    // 1) qkv = hidden @ qkv_w^T + qkv_b   (bf16 out)
    gemm_bias_kernel<128, float, unsigned short>
        <<<dim3(QKVN / 128, SEQ / 128), 256, 0, stream>>>(
            hidden, HID, qkv_w, HID, qkv_b, qkv, QKVN, SEQ, QKVN, HID);

    // 2) RoPE + head split / V transpose
    rope_reshape_kernel<<<(SEQ * 80 * 32) / 256, 256, 0, stream>>>(qkv, qh, kh, vT);

    // 3) windowed attention with sinks
    attn_kernel<<<dim3(SEQ / 64, NH), 128, 0, stream>>>(qh, kh, vT, sinks, attnout);

    // 4) out = attnout @ out_w^T + out_b  (f32 out)
    gemm_bias_kernel<64, unsigned short, float>
        <<<dim3(HID / 64, SEQ / 128), 256, 0, stream>>>(
            attnout, AOUTK, out_w, AOUTK, out_b, out, HID, SEQ, HID, AOUTK);
}
